// Block_32882269618638
// MI455X (gfx1250) — matrix operand order repaired and checked by reading, not compiled, not run on hardware
//
#include <hip/hip_runtime.h>
#include <hip/hip_bf16.h>
#include <math.h>

// ---------------------------------------------------------------------------
// Types for CDNA5 WMMA (wave32, v_wmma_f32_16x16x32_bf16)
// ---------------------------------------------------------------------------
typedef __bf16 bf16_t;
typedef __attribute__((ext_vector_type(16))) __bf16 v16bf;
typedef __attribute__((ext_vector_type(8)))  __bf16 v8bf;
typedef __attribute__((ext_vector_type(8)))  float  v8f;
typedef __attribute__((ext_vector_type(4)))  int    v4i;

union frag16 { v16bf v; v8bf h[2]; };

// Async global->LDS staging (CDNA5 GLOBAL_LOAD_ASYNC_TO_LDS_B128, ASYNCcnt).
// Guarded so host pass / toolchains without the builtin fall back cleanly.
#if defined(__AMDGCN__) && \
    __has_builtin(__builtin_amdgcn_global_load_async_to_lds_b128) && \
    __has_builtin(__builtin_amdgcn_s_wait_asynccnt)
#define USE_ASYNC_LDS 1
#else
#define USE_ASYNC_LDS 0
#endif

#if USE_ASYNC_LDS
typedef __attribute__((address_space(1))) v4i* gas_ptr;   // global int4*
typedef __attribute__((address_space(3))) v4i* las_ptr;   // LDS int4*
__device__ __forceinline__ void async_copy16(const void* g, void* l) {
  // generic LDS address truncates to the 32-bit LDS offset (ISA 10.2 aperture)
  __builtin_amdgcn_global_load_async_to_lds_b128(
      (gas_ptr)(size_t)g, (las_ptr)(unsigned)(size_t)l, 0, 0);
}
__device__ __forceinline__ void async_wait0() {
  __builtin_amdgcn_s_wait_asynccnt(0);
}
#else
__device__ __forceinline__ void async_wait0() {}
#endif

__device__ __forceinline__ bf16_t f2bf(float f) {
  unsigned u = __float_as_uint(f);
  unsigned r = (u + 0x7FFFu + ((u >> 16) & 1u)) >> 16;   // RNE
  return __builtin_bit_cast(bf16_t, (unsigned short)r);
}

__device__ __forceinline__ float gelu_tanh(float x) {
  const float c = 0.7978845608028654f;
  float x3 = x * x * x;
  return 0.5f * x * (1.0f + tanhf(c * (x + 0.044715f * x3)));
}

// ---------------------------------------------------------------------------
// Weight cast+transpose: Wt[n][k] = bf16(W[k][n]).  32x32 tiles via LDS.
// ---------------------------------------------------------------------------
__global__ __launch_bounds__(256) void cast_transpose_kernel(
    const float* __restrict__ W, bf16_t* __restrict__ Wt, int K, int N) {
  __shared__ float s[32][33];
  int tx = threadIdx.x & 31, ty = threadIdx.x >> 5;
  int n0 = blockIdx.x * 32, k0 = blockIdx.y * 32;
#pragma unroll
  for (int j = 0; j < 4; ++j) {
    int k = k0 + ty + j * 8;
    s[ty + j * 8][tx] = W[(size_t)k * N + n0 + tx];
  }
  __syncthreads();
#pragma unroll
  for (int j = 0; j < 4; ++j) {
    int n = n0 + ty + j * 8;
    Wt[(size_t)n * K + k0 + tx] = f2bf(s[tx][ty + j * 8]);
  }
}

// ---------------------------------------------------------------------------
// LayerNorm over C=1024, one row per block (256 threads), bf16 output.
// ---------------------------------------------------------------------------
__global__ __launch_bounds__(256) void ln_kernel(
    const float* __restrict__ x, const float* __restrict__ g,
    const float* __restrict__ b, bf16_t* __restrict__ out) {
  const int C = 1024;
  int row = blockIdx.x;
  const float* xr = x + (size_t)row * C;
  int t = threadIdx.x;
  float4 v4 = *(const float4*)(xr + t * 4);
  float v[4] = {v4.x, v4.y, v4.z, v4.w};
  float s = 0.f, ss = 0.f;
#pragma unroll
  for (int i = 0; i < 4; ++i) { s += v[i]; ss += v[i] * v[i]; }
#pragma unroll
  for (int off = 16; off >= 1; off >>= 1) {
    s += __shfl_xor(s, off, 32);
    ss += __shfl_xor(ss, off, 32);
  }
  __shared__ float sm[8], sm2[8];
  int w = t >> 5, lane = t & 31;
  if (lane == 0) { sm[w] = s; sm2[w] = ss; }
  __syncthreads();
  if (w == 0) {
    float a  = (lane < 8) ? sm[lane]  : 0.f;
    float a2 = (lane < 8) ? sm2[lane] : 0.f;
#pragma unroll
    for (int off = 4; off >= 1; off >>= 1) {
      a += __shfl_xor(a, off, 32);
      a2 += __shfl_xor(a2, off, 32);
    }
    if (lane == 0) { sm[0] = a; sm2[0] = a2; }
  }
  __syncthreads();
  float mu  = sm[0] * (1.0f / C);
  float var = sm2[0] * (1.0f / C) - mu * mu;
  float rstd = rsqrtf(var + 1e-5f);
  bf16_t* orow = out + (size_t)row * C;
#pragma unroll
  for (int i = 0; i < 4; ++i) {
    int c = t * 4 + i;
    orow[c] = f2bf((v[i] - mu) * rstd * g[c] + b[c]);
  }
}

// ---------------------------------------------------------------------------
// Tiled bf16 WMMA GEMM: D[M,N] = epi(A[M,K] @ Wt[N,K]^T + bias, resid)
// 128x128 tile, BLOCK_K=32, 256 threads = 8 waves, wave tile 32x64.
// EPI: 0 = bias -> bf16 | 1 = gelu(bias) -> bf16 | 2 = bias+resid -> f32
// ---------------------------------------------------------------------------
template <int EPI>
__global__ __launch_bounds__(256) void gemm_bf16_kernel(
    const bf16_t* __restrict__ A, const bf16_t* __restrict__ Bt,
    const float* __restrict__ bias, const float* __restrict__ resid,
    void* __restrict__ dst, int M, int N, int K) {
  __shared__ bf16_t As[2][128][32];
  __shared__ bf16_t Bs[2][128][32];

  int t = threadIdx.x, w = t >> 5, lane = t & 31;
  int hi16 = lane >> 4, l16 = lane & 15;
  int m0 = blockIdx.y * 128, n0 = blockIdx.x * 128;
  int wm0 = (w >> 1) * 32, wn0 = (w & 1) * 64;

  v8f acc[2][4];
#pragma unroll
  for (int i = 0; i < 2; ++i)
#pragma unroll
    for (int j = 0; j < 4; ++j)
#pragma unroll
      for (int e = 0; e < 8; ++e) acc[i][j][e] = 0.f;

  int lrow = t >> 1;           // 0..127
  int lc16 = (t & 1) * 16;     // 0 or 16

  auto loadTile = [&](int kk, int buf) {
    const bf16_t* ap = A + (size_t)(m0 + lrow) * K + kk + lc16;
    const bf16_t* bp = Bt + (size_t)(n0 + lrow) * K + kk + lc16;
#if USE_ASYNC_LDS
    async_copy16(ap,     &As[buf][lrow][lc16]);
    async_copy16(ap + 8, &As[buf][lrow][lc16 + 8]);
    async_copy16(bp,     &Bs[buf][lrow][lc16]);
    async_copy16(bp + 8, &Bs[buf][lrow][lc16 + 8]);
#else
    *(v8bf*)&As[buf][lrow][lc16]     = *(const v8bf*)(ap);
    *(v8bf*)&As[buf][lrow][lc16 + 8] = *(const v8bf*)(ap + 8);
    *(v8bf*)&Bs[buf][lrow][lc16]     = *(const v8bf*)(bp);
    *(v8bf*)&Bs[buf][lrow][lc16 + 8] = *(const v8bf*)(bp + 8);
#endif
  };

  loadTile(0, 0);
  async_wait0();
  __syncthreads();

  int nk = K / 32;
  for (int ks = 0; ks < nk; ++ks) {
    int buf = ks & 1;
    if (ks + 1 < nk) loadTile((ks + 1) * 32, buf ^ 1);
    if (ks + 2 < nk) {
      __builtin_prefetch((const void*)(A + (size_t)(m0 + lrow) * K + (ks + 2) * 32), 0, 0);
      __builtin_prefetch((const void*)(Bt + (size_t)(n0 + lrow) * K + (ks + 2) * 32), 0, 0);
    }

    frag16 af[2], bfm[4];
    int kb = hi16 ? 16 : 0;      // A layout: lanes<16 K0..15, lanes>=16 K16..31
#pragma unroll
    for (int i = 0; i < 2; ++i) {
      int row = wm0 + i * 16 + l16;
      af[i].h[0] = *(const v8bf*)&As[buf][row][kb];
      af[i].h[1] = *(const v8bf*)&As[buf][row][kb + 8];
    }
    int koff = hi16 ? 16 : 0;    // B layout: lanes<16 K0..15, lanes>=16 K16..31
#pragma unroll
    for (int j = 0; j < 4; ++j) {
      int col = wn0 + j * 16 + l16;
      bfm[j].h[0] = *(const v8bf*)&Bs[buf][col][koff];
      bfm[j].h[1] = *(const v8bf*)&Bs[buf][col][koff + 8];
    }
#pragma unroll
    for (int i = 0; i < 2; ++i)
#pragma unroll
      for (int j = 0; j < 4; ++j)
        acc[i][j] = __builtin_amdgcn_wmma_f32_16x16x32_bf16(
            false, af[i].v, false, bfm[j].v, (short)0, acc[i][j], false, false);
    async_wait0();
    __syncthreads();
  }

  // Epilogue: C/D layout — VGPR r: lanes 0-15 -> M=r, lanes 16-31 -> M=r+8, N=lane&15
#pragma unroll
  for (int i = 0; i < 2; ++i) {
#pragma unroll
    for (int j = 0; j < 4; ++j) {
#pragma unroll
      for (int r = 0; r < 8; ++r) {
        int m = m0 + wm0 + i * 16 + r + hi16 * 8;
        int n = n0 + wn0 + j * 16 + l16;
        float v = acc[i][j][r] + bias[n];
        if (EPI == 1) v = gelu_tanh(v);
        if (EPI == 2) {
          v += resid[(size_t)m * N + n];
          ((float*)dst)[(size_t)m * N + n] = v;
        } else {
          ((bf16_t*)dst)[(size_t)m * N + n] = f2bf(v);
        }
      }
    }
  }
}

// ---------------------------------------------------------------------------
// Flash attention: causal, per (q-tile 64, head, batch). 4 waves x 16 q-rows.
// qkv: [B*T, 3C] bf16 (q|k|v).  y: [B*T, C] bf16.
// ---------------------------------------------------------------------------
__global__ __launch_bounds__(128) void attn_kernel(
    const bf16_t* __restrict__ qkv, bf16_t* __restrict__ y, int T) {
  const int C = 1024, QKV = 3072;
  int qt = blockIdx.x, hh = blockIdx.y, bb = blockIdx.z;
  int t = threadIdx.x, w = t >> 5, lane = t & 31;
  int hi16 = lane >> 4, l16 = lane & 15;
  size_t base = (size_t)bb * T;

  const bf16_t* Qp = qkv + 0    + hh * 64;
  const bf16_t* Kp = qkv + 1024 + hh * 64;
  const bf16_t* Vp = qkv + 2048 + hh * 64;

  __shared__ bf16_t Kl[64][64];
  __shared__ bf16_t Vl[64][64];       // Vl[d][k] (transposed)
  __shared__ bf16_t Pl[4][16][64];    // per-wave P tile

  // Q fragments for this wave's 16 rows (K-dim = D = 64 -> 2 chunks of 32)
  int qrow = qt * 64 + w * 16 + l16;
  frag16 aq[2];
  {
    const bf16_t* qp = Qp + (base + qrow) * QKV;
    int kb = hi16 ? 16 : 0;
    aq[0].h[0] = *(const v8bf*)(qp + kb);
    aq[0].h[1] = *(const v8bf*)(qp + kb + 8);
    aq[1].h[0] = *(const v8bf*)(qp + 32 + kb);
    aq[1].h[1] = *(const v8bf*)(qp + 32 + kb + 8);
  }

  v8f o[4];
#pragma unroll
  for (int j = 0; j < 4; ++j)
#pragma unroll
    for (int e = 0; e < 8; ++e) o[j][e] = 0.f;
  float mrow[8], lrow[8];
#pragma unroll
  for (int r = 0; r < 8; ++r) { mrow[r] = -3.0e38f; lrow[r] = 0.f; }

  for (int kt = 0; kt <= qt; ++kt) {
    // stage K (row-major, async when available) and V (transposed) tiles in LDS
#pragma unroll
    for (int i = 0; i < 4; ++i) {
      int chunk = t + i * 128;                 // 512 chunks of 8 elems
      int row = chunk >> 3, c8 = (chunk & 7) * 8;
      const bf16_t* kp = Kp + (base + kt * 64 + row) * QKV + c8;
#if USE_ASYNC_LDS
      async_copy16(kp, &Kl[row][c8]);
#else
      *(v8bf*)&Kl[row][c8] = *(const v8bf*)kp;
#endif
      v8bf vv = *(const v8bf*)(Vp + (base + kt * 64 + row) * QKV + c8);
#pragma unroll
      for (int e = 0; e < 8; ++e) Vl[c8 + e][row] = vv[e];
    }
    async_wait0();
    __syncthreads();

    // S = (Q K^T) * 1/sqrt(D), causal mask
    v8f sv[4];
#pragma unroll
    for (int j = 0; j < 4; ++j) {
      v8f s;
#pragma unroll
      for (int e = 0; e < 8; ++e) s[e] = 0.f;
#pragma unroll
      for (int kc = 0; kc < 2; ++kc) {
        frag16 bk;
        int n = j * 16 + l16;
        int koff = kc * 32 + (hi16 ? 16 : 0);
        bk.h[0] = *(const v8bf*)&Kl[n][koff];
        bk.h[1] = *(const v8bf*)&Kl[n][koff + 8];
        s = __builtin_amdgcn_wmma_f32_16x16x32_bf16(
            false, aq[kc].v, false, bk.v, (short)0, s, false, false);
      }
#pragma unroll
      for (int r = 0; r < 8; ++r) {
        int qg = qt * 64 + w * 16 + r + hi16 * 8;
        int kg = kt * 64 + j * 16 + l16;
        float sc = s[r] * 0.125f;
        s[r] = (kg <= qg) ? sc : -3.0e38f;
      }
      sv[j] = s;
    }

    // online softmax (row reductions across 16-lane halves)
#pragma unroll
    for (int r = 0; r < 8; ++r) {
      float lm = fmaxf(fmaxf(sv[0][r], sv[1][r]), fmaxf(sv[2][r], sv[3][r]));
#pragma unroll
      for (int off = 8; off >= 1; off >>= 1) lm = fmaxf(lm, __shfl_xor(lm, off, 32));
      float mnew = fmaxf(mrow[r], lm);
      float scale = __expf(mrow[r] - mnew);
      float rs = 0.f;
#pragma unroll
      for (int j = 0; j < 4; ++j) {
        float p = __expf(sv[j][r] - mnew);
        sv[j][r] = p;
        rs += p;
      }
#pragma unroll
      for (int off = 8; off >= 1; off >>= 1) rs += __shfl_xor(rs, off, 32);
      lrow[r] = lrow[r] * scale + rs;
      mrow[r] = mnew;
#pragma unroll
      for (int j = 0; j < 4; ++j) o[j][r] *= scale;
    }

    // P -> LDS (accumulator layout -> memory) for A-fragment reuse
#pragma unroll
    for (int j = 0; j < 4; ++j)
#pragma unroll
      for (int r = 0; r < 8; ++r)
        Pl[w][r + hi16 * 8][j * 16 + l16] = f2bf(sv[j][r]);

    // O += P @ V  (K-dim = 64 k-cols -> 2 chunks)
#pragma unroll
    for (int kc = 0; kc < 2; ++kc) {
      frag16 ap;
      int kb = kc * 32 + (hi16 ? 16 : 0);
      ap.h[0] = *(const v8bf*)&Pl[w][l16][kb];
      ap.h[1] = *(const v8bf*)&Pl[w][l16][kb + 8];
#pragma unroll
      for (int j = 0; j < 4; ++j) {
        frag16 bv;
        int n = j * 16 + l16;
        int koff = kc * 32 + (hi16 ? 16 : 0);
        bv.h[0] = *(const v8bf*)&Vl[n][koff];
        bv.h[1] = *(const v8bf*)&Vl[n][koff + 8];
        o[j] = __builtin_amdgcn_wmma_f32_16x16x32_bf16(
            false, ap.v, false, bv.v, (short)0, o[j], false, false);
      }
    }
    __syncthreads();
  }

  // normalize and store y (bf16)
#pragma unroll
  for (int r = 0; r < 8; ++r) {
    float inv = 1.0f / lrow[r];
    int qg = qt * 64 + w * 16 + r + hi16 * 8;
    bf16_t* yp = y + (base + qg) * C + hh * 64;
#pragma unroll
    for (int j = 0; j < 4; ++j) yp[j * 16 + l16] = f2bf(o[j][r] * inv);
  }
}

// ---------------------------------------------------------------------------
// Host-side orchestration
// ---------------------------------------------------------------------------
extern "C" void kernel_launch(void* const* d_in, const int* in_sizes, int n_in,
                              void* d_out, int out_size, void* d_ws, size_t ws_size,
                              hipStream_t stream) {
  const float* x       = (const float*)d_in[0];
  const float* ln1_g   = (const float*)d_in[1];
  const float* ln1_b   = (const float*)d_in[2];
  const float* w_attn  = (const float*)d_in[3];
  const float* b_attn  = (const float*)d_in[4];
  const float* w_proj  = (const float*)d_in[5];
  const float* b_proj  = (const float*)d_in[6];
  const float* ln2_g   = (const float*)d_in[7];
  const float* ln2_b   = (const float*)d_in[8];
  const float* w_fc    = (const float*)d_in[9];
  const float* b_fc    = (const float*)d_in[10];
  const float* w_proj2 = (const float*)d_in[11];
  const float* b_proj2 = (const float*)d_in[12];
  float* out = (float*)d_out;

  const int Bn = 2, T = 2048, C = 1024;
  const int M = Bn * T;  // 4096

  char* ws = (char*)d_ws;
  size_t off = 0;
  auto alloc = [&](size_t bytes) -> void* {
    void* p = ws + off;
    off = (off + bytes + 255) & ~(size_t)255;
    return p;
  };
  bf16_t* wt_attn  = (bf16_t*)alloc((size_t)3 * C * C * 2);      // [3C][C]
  bf16_t* wt_proj  = (bf16_t*)alloc((size_t)C * C * 2);          // [C][C]
  bf16_t* wt_fc    = (bf16_t*)alloc((size_t)4 * C * C * 2);      // [4C][C]
  bf16_t* wt_proj2 = (bf16_t*)alloc((size_t)4 * C * C * 2);      // [C][4C]
  bf16_t* h1   = (bf16_t*)alloc((size_t)M * C * 2);
  bf16_t* qkv  = (bf16_t*)alloc((size_t)M * 3 * C * 2);
  bf16_t* yatt = (bf16_t*)alloc((size_t)M * C * 2);
  float*  x1   = (float*)alloc((size_t)M * C * 4);
  bf16_t* h2   = (bf16_t*)alloc((size_t)M * C * 2);
  bf16_t* fca  = (bf16_t*)alloc((size_t)M * 4 * C * 2);

  // 1) weights -> bf16, transposed to [N][K]
  cast_transpose_kernel<<<dim3(3 * C / 32, C / 32), 256, 0, stream>>>(w_attn, wt_attn, C, 3 * C);
  cast_transpose_kernel<<<dim3(C / 32, C / 32), 256, 0, stream>>>(w_proj, wt_proj, C, C);
  cast_transpose_kernel<<<dim3(4 * C / 32, C / 32), 256, 0, stream>>>(w_fc, wt_fc, C, 4 * C);
  cast_transpose_kernel<<<dim3(C / 32, 4 * C / 32), 256, 0, stream>>>(w_proj2, wt_proj2, 4 * C, C);

  // 2) h1 = ln1(x)
  ln_kernel<<<M, 256, 0, stream>>>(x, ln1_g, ln1_b, h1);

  // 3) qkv = h1 @ w_attn + b_attn   (bf16 out)
  gemm_bf16_kernel<0><<<dim3(3 * C / 128, M / 128), 256, 0, stream>>>(
      h1, wt_attn, b_attn, nullptr, (void*)qkv, M, 3 * C, C);

  // 4) y = causal_attention(qkv)
  attn_kernel<<<dim3(T / 64, 16, Bn), 128, 0, stream>>>(qkv, yatt, T);

  // 5) x1 = x + y @ w_proj + b_proj   (f32 out)
  gemm_bf16_kernel<2><<<dim3(C / 128, M / 128), 256, 0, stream>>>(
      yatt, wt_proj, b_proj, x, (void*)x1, M, C, C);

  // 6) h2 = ln2(x1)
  ln_kernel<<<M, 256, 0, stream>>>(x1, ln2_g, ln2_b, h2);

  // 7) fca = gelu(h2 @ w_fc + b_fc)   (bf16 out)
  gemm_bf16_kernel<1><<<dim3(4 * C / 128, M / 128), 256, 0, stream>>>(
      h2, wt_fc, b_fc, nullptr, (void*)fca, M, 4 * C, C);

  // 8) out = x1 + fca @ w_proj2 + b_proj2   (f32 out)
  gemm_bf16_kernel<2><<<dim3(C / 128, M / 128), 256, 0, stream>>>(
      fca, wt_proj2, b_proj2, x1, (void*)out, M, C, 4 * C);
}
